// DeformRoIPoolPackWraper_1202590843764
// MI455X (gfx1250) — matrix operand order repaired and checked by reading, not compiled, not run on hardware
//
#include <hip/hip_runtime.h>

// ---------------------------------------------------------------------------
// DeformRoIPoolPack for MI455X (gfx1250, wave32)
// pool1 -> FC1 (WMMA f16/f32-acc, async-to-LDS double-buffered) + ReLU
//       -> FC2 -> pool2
// ---------------------------------------------------------------------------

typedef __attribute__((ext_vector_type(16))) _Float16 v16h;
typedef __attribute__((ext_vector_type(8)))  _Float16 h8;
typedef __attribute__((ext_vector_type(8)))  float    v8f;

#define FEAT_B 2
#define FEAT_C 256
#define FEAT_H 112
#define FEAT_W 112
#define N_ROIS 1024
#define OUT_HW 49          // 7*7
#define FC_IN  12544       // 256*49
#define DFC    1024
#define FC2_OUT 98         // 2*49
#define GAMMA_F 0.1f
#define SCALE_F 0.25f

// ---------------------------------------------------------------------------
// 1) NCHW -> NHWC transpose of the feature map (makes pool gathers coalesced)
// ---------------------------------------------------------------------------
__global__ void nchw_to_nhwc_kernel(const float* __restrict__ in, float* __restrict__ outp) {
    int idx = blockIdx.x * 256 + threadIdx.x;              // b*HW*C + hw*C + c
    int c    = idx & (FEAT_C - 1);
    int rest = idx >> 8;                                   // b*HW + hw
    int hw   = rest % (FEAT_H * FEAT_W);
    int b    = rest / (FEAT_H * FEAT_W);
    outp[idx] = in[((size_t)(b * FEAT_C + c)) * (FEAT_H * FEAT_W) + hw];
}

// ---------------------------------------------------------------------------
// 2) w1 [K,N] f32  ->  W1t [N,K] f16   (LDS-tiled transpose, coalesced I/O)
//    grid: (N/32, K/32) = (32, 392), block: 256 = 32x8
// ---------------------------------------------------------------------------
__global__ void w1_transpose_cvt_kernel(const float* __restrict__ src, _Float16* __restrict__ dst) {
    __shared__ float tile[32][33];
    int n0 = blockIdx.x * 32;
    int k0 = blockIdx.y * 32;
    int tx = threadIdx.x & 31;
    int ty = threadIdx.x >> 5;                 // 0..7
#pragma unroll
    for (int i = 0; i < 4; ++i) {
        int k = ty + i * 8;
        tile[k][tx] = src[(size_t)(k0 + k) * DFC + n0 + tx];   // coalesced read
    }
    __syncthreads();
#pragma unroll
    for (int i = 0; i < 4; ++i) {
        int n = ty + i * 8;
        dst[(size_t)(n0 + n) * FC_IN + k0 + tx] = (_Float16)tile[tx][n];  // coalesced write
    }
}

// ---------------------------------------------------------------------------
// Deformable RoI pooling core (sampling_ratio = 2, bilinear, NHWC gathers)
// ---------------------------------------------------------------------------
__device__ __forceinline__ float pool_bin(const float* __restrict__ fhwc, int b, int c,
                                          float ys, float xs, float bh, float bw,
                                          int ph, int pw) {
    const int H = FEAT_H, W = FEAT_W, C = FEAT_C;
    const float* base = fhwc + ((size_t)b * H * W) * C + c;
    float sum = 0.f;
#pragma unroll
    for (int iy = 0; iy < 2; ++iy) {
        float y = ys + bh * ((float)ph + (iy + 0.5f) * 0.5f);
#pragma unroll
        for (int ix = 0; ix < 2; ++ix) {
            float x = xs + bw * ((float)pw + (ix + 0.5f) * 0.5f);
            if (!(y > -1.0f && y < (float)H && x > -1.0f && x < (float)W)) continue;
            float yc = fminf(fmaxf(y, 0.f), (float)(H - 1));
            float xc = fminf(fmaxf(x, 0.f), (float)(W - 1));
            float y0f = floorf(yc), x0f = floorf(xc);
            float ly = yc - y0f, lx = xc - x0f;
            float hy = 1.f - ly,  hx = 1.f - lx;
            int y0 = (int)y0f, x0 = (int)x0f;
            int y1 = min(y0 + 1, H - 1), x1 = min(x0 + 1, W - 1);
            float v00 = base[(size_t)(y0 * W + x0) * C];
            float v01 = base[(size_t)(y0 * W + x1) * C];
            float v10 = base[(size_t)(y1 * W + x0) * C];
            float v11 = base[(size_t)(y1 * W + x1) * C];
            sum += hy * hx * v00 + hy * lx * v01 + ly * hx * v10 + ly * lx * v11;
        }
    }
    return sum * 0.25f;   // / (g*g)
}

// grid: (49 bins, 1024 rois), block: 256 threads = channels
template <bool WITH_OFFS, bool TO_HALF>
__global__ void pool_kernel(const float* __restrict__ fhwc, const float* __restrict__ rois,
                            const float* __restrict__ offs, _Float16* __restrict__ outH,
                            float* __restrict__ outF) {
    int bin = blockIdx.x;            // ph*7+pw
    int n   = blockIdx.y;
    int c   = threadIdx.x;
    int ph  = bin / 7, pw = bin % 7;

    const float* r = rois + n * 5;
    int   b  = (int)r[0];
    float x1 = r[1] * SCALE_F - 0.5f;
    float y1 = r[2] * SCALE_F - 0.5f;
    float x2 = r[3] * SCALE_F - 0.5f;
    float y2 = r[4] * SCALE_F - 0.5f;
    float rw = x2 - x1, rh = y2 - y1;
    float bw = rw * (1.f / 7.f), bh = rh * (1.f / 7.f);
    float xs = x1, ys = y1;
    if (WITH_OFFS) {
        xs += GAMMA_F * rw * offs[n * FC2_OUT + bin];
        ys += GAMMA_F * rh * offs[n * FC2_OUT + OUT_HW + bin];
    }
    float v = pool_bin(fhwc, b, c, ys, xs, bh, bw, ph, pw);
    if (TO_HALF) {
        // GEMM A layout: X[roi][c*49 + bin]  (== x.reshape(N,-1))
        outH[(size_t)n * FC_IN + c * OUT_HW + bin] = (_Float16)v;
    } else {
        // final output: (N, C, 7, 7)
        outF[((size_t)n * FEAT_C + c) * OUT_HW + bin] = v;
    }
}

// ---------------------------------------------------------------------------
// 3) GEMM1: H[1024,1024] = relu(X[1024,12544] @ W1[12544,1024] + b1)
//    A = X [M,K] row-major f16, Bt = W1^T [N,K] row-major f16.
//    128x128 block tile, 8 waves (wave tile 32x64 = 2x4 WMMA tiles),
//    K in 32-wide slices, double-buffered GLOBAL_LOAD_ASYNC_TO_LDS_B128.
// ---------------------------------------------------------------------------
__global__ __launch_bounds__(256) void gemm1_kernel(const _Float16* __restrict__ A,
                                                    const _Float16* __restrict__ Bt,
                                                    const float* __restrict__ bias,
                                                    float* __restrict__ Hout) {
    __shared__ _Float16 As[2][128][40];   // [buf][row][k], 80B row stride (16B aligned)
    __shared__ _Float16 Bs[2][128][40];   // [buf][col][k]

    const int tid  = threadIdx.x;
    const int lane = tid & 31;
    const int wave = tid >> 5;
    const int lr   = lane & 15;        // row (A) / col (B) / N (C,D)
    const int hi   = lane >> 4;        // half-wave selector
    const int wrow = wave & 3;         // 4 wave-rows  * 32 M
    const int wcol = wave >> 2;        // 2 wave-cols  * 64 N
    const int M0 = blockIdx.y * 128, N0 = blockIdx.x * 128;

    v8f acc[2][4];
#pragma unroll
    for (int mt = 0; mt < 2; ++mt)
#pragma unroll
        for (int nt = 0; nt < 4; ++nt)
#pragma unroll
            for (int e = 0; e < 8; ++e) acc[mt][nt][e] = 0.f;

    // staging assignment: thread -> one 32B row-chunk of each tile
    const int sr  = tid >> 1;            // 128 rows, 2 threads/row
    const int skp = (tid & 1) << 4;      // half offset 0 / 16
    const _Float16* aRow = A  + (size_t)(M0 + sr) * FC_IN + skp;
    const _Float16* bRow = Bt + (size_t)(N0 + sr) * FC_IN + skp;

    // async global->LDS stage of one 128x32 A tile + 128x32 B tile
    auto stage = [&](int buf, int k0) {
        unsigned int la = (unsigned int)(size_t)&As[buf][sr][skp];
        unsigned int lb = (unsigned int)(size_t)&Bs[buf][sr][skp];
        unsigned long long ga = (unsigned long long)(size_t)(aRow + k0);
        unsigned long long gb = (unsigned long long)(size_t)(bRow + k0);
        asm volatile(
            "global_load_async_to_lds_b128 %0, %2, off\n\t"
            "global_load_async_to_lds_b128 %0, %2, off offset:16\n\t"
            "global_load_async_to_lds_b128 %1, %3, off\n\t"
            "global_load_async_to_lds_b128 %1, %3, off offset:16"
            :: "v"(la), "v"(lb), "v"(ga), "v"(gb) : "memory");
    };

    stage(0, 0);
    for (int k0 = 0, it = 0; k0 < FC_IN; k0 += 32, ++it) {
        const int cur = it & 1;
        const int nxt = cur ^ 1;
        if (k0 + 32 < FC_IN) {
            stage(nxt, k0 + 32);   // prefetch next slice (buffer was drained last iter)
            // 4 async ops/wave per stage; in-order completion => <=4 means current done
            asm volatile("s_wait_asynccnt 0x4" ::: "memory");
        } else {
            asm volatile("s_wait_asynccnt 0x0" ::: "memory");
        }
        __syncthreads();

        // fragments per ISA wave32 layouts
        union { v16h v; h8 h[2]; } af[2], bf[4];
#pragma unroll
        for (int mt = 0; mt < 2; ++mt) {
            int row = wrow * 32 + mt * 16 + lr;
            af[mt].h[0] = *(const h8*)&As[cur][row][hi * 8];       // K 0..7  (+8 hi lanes)
            af[mt].h[1] = *(const h8*)&As[cur][row][16 + hi * 8];  // K 16..23 (+8)
        }
#pragma unroll
        for (int nt = 0; nt < 4; ++nt) {
            int col = wcol * 64 + nt * 16 + lr;
            bf[nt].h[0] = *(const h8*)&Bs[cur][col][hi * 8];       // K 0..7  (+8 hi lanes)
            bf[nt].h[1] = *(const h8*)&Bs[cur][col][16 + hi * 8];  // K 16..23 (+8)
        }
#pragma unroll
        for (int mt = 0; mt < 2; ++mt)
#pragma unroll
            for (int nt = 0; nt < 4; ++nt)
                acc[mt][nt] = __builtin_amdgcn_wmma_f32_16x16x32_f16(
                    false, af[mt].v, false, bf[nt].v, (short)0, acc[mt][nt], false, false);
        __syncthreads();   // all reads of 'cur' done before it becomes async target
    }

    // epilogue: bias + ReLU, f32 out. D layout: vgpr j -> M = j + 8*hi, N = lr
#pragma unroll
    for (int mt = 0; mt < 2; ++mt) {
#pragma unroll
        for (int nt = 0; nt < 4; ++nt) {
            int col = N0 + wcol * 64 + nt * 16 + lr;
            float bv = bias[col];
#pragma unroll
            for (int j = 0; j < 8; ++j) {
                int row = M0 + wrow * 32 + mt * 16 + j + 8 * hi;
                float v = acc[mt][nt][j] + bv;
                Hout[(size_t)row * DFC + col] = v > 0.f ? v : 0.f;
            }
        }
    }
}

// ---------------------------------------------------------------------------
// 4) GEMM2: offs[1024,98] = H[1024,1024] @ W2[1024,98] + b2  (tiny, f32 VALU)
// ---------------------------------------------------------------------------
__global__ void gemm2_kernel(const float* __restrict__ H, const float* __restrict__ W2,
                             const float* __restrict__ b2, float* __restrict__ offs) {
    int n = blockIdx.x;
    int t = threadIdx.x;
    if (t >= FC2_OUT) return;
    const float* h = H + (size_t)n * DFC;
    float acc = b2[t];
#pragma unroll 4
    for (int k = 0; k < DFC; ++k) acc += h[k] * W2[(size_t)k * FC2_OUT + t];
    offs[(size_t)n * FC2_OUT + t] = acc;
}

// ---------------------------------------------------------------------------
// Orchestration
// ---------------------------------------------------------------------------
extern "C" void kernel_launch(void* const* d_in, const int* in_sizes, int n_in,
                              void* d_out, int out_size, void* d_ws, size_t ws_size,
                              hipStream_t stream) {
    (void)in_sizes; (void)n_in; (void)out_size; (void)ws_size;
    const float* feat = (const float*)d_in[0];
    const float* rois = (const float*)d_in[1];
    const float* w1   = (const float*)d_in[2];
    const float* b1   = (const float*)d_in[3];
    const float* w2   = (const float*)d_in[4];
    const float* b2   = (const float*)d_in[5];
    float* out = (float*)d_out;

    char* ws = (char*)d_ws;
    size_t off = 0;
    auto take = [&](size_t bytes) -> void* {
        void* p = ws + off;
        off += (bytes + 255) & ~(size_t)255;
        return p;
    };
    float*    fhwc = (float*)   take((size_t)FEAT_B * FEAT_H * FEAT_W * FEAT_C * 4);
    _Float16* Xh   = (_Float16*)take((size_t)N_ROIS * FC_IN * 2);
    _Float16* W1t  = (_Float16*)take((size_t)FC_IN * DFC * 2);     // [N][K]
    float*    Hbuf = (float*)   take((size_t)N_ROIS * DFC * 4);
    float*    offs = (float*)   take((size_t)N_ROIS * FC2_OUT * 4);

    // 1) feature map NCHW -> NHWC
    {
        int n = FEAT_B * FEAT_H * FEAT_W * FEAT_C;           // 6,422,528 (multiple of 256)
        nchw_to_nhwc_kernel<<<dim3(n / 256), dim3(256), 0, stream>>>(feat, fhwc);
    }
    // 2) w1 -> f16, transposed to [N][K] so GEMM B staging is contiguous
    w1_transpose_cvt_kernel<<<dim3(DFC / 32, FC_IN / 32), dim3(256), 0, stream>>>(w1, W1t);
    // 3) pool1 (no offsets) -> f16 GEMM-A matrix
    pool_kernel<false, true><<<dim3(OUT_HW, N_ROIS), dim3(256), 0, stream>>>(
        fhwc, rois, (const float*)nullptr, Xh, (float*)nullptr);
    // 4) FC1 + ReLU via WMMA (async double-buffered)
    gemm1_kernel<<<dim3(DFC / 128, N_ROIS / 128), dim3(256), 0, stream>>>(Xh, W1t, b1, Hbuf);
    // 5) FC2 -> offsets
    gemm2_kernel<<<dim3(N_ROIS), dim3(128), 0, stream>>>(Hbuf, w2, b2, offs);
    // 6) pool2 (with offsets) -> final output
    pool_kernel<true, false><<<dim3(OUT_HW, N_ROIS), dim3(256), 0, stream>>>(
        fhwc, rois, offs, (_Float16*)nullptr, out);
}
